// FD_4612794876084
// MI455X (gfx1250) — matrix operand order repaired and checked by reading, not compiled, not run on hardware
//
#include <hip/hip_runtime.h>
#include <hip/hip_bf16.h>
#include <math.h>

// ---------------------------------------------------------------------------
// Types for CDNA5 WMMA (wave32): v_wmma_f32_16x16x32_bf16
// ---------------------------------------------------------------------------
typedef __bf16 bf16x16 __attribute__((ext_vector_type(16)));
typedef __bf16 bf16x8  __attribute__((ext_vector_type(8)));
typedef float  f32x8   __attribute__((ext_vector_type(8)));
typedef int    v4i_    __attribute__((ext_vector_type(4)));

union Frag16 { bf16x16 v; bf16x8 h[2]; };

#define ODE_STEPS 6

// Async global->LDS path (gfx1250): guarded so the kernel compiles either way.
#if defined(__has_builtin)
#if __has_builtin(__builtin_amdgcn_global_load_async_to_lds_b128) && \
    __has_builtin(__builtin_amdgcn_s_wait_asynccnt)
#define HAVE_ASYNC_LDS 1
#endif
#endif
#ifndef HAVE_ASYNC_LDS
#define HAVE_ASYNC_LDS 0
#endif

// ---------------------------------------------------------------------------
// Pack f32 row-major [M,K] into bf16 WMMA A-fragments.
// A-fragment k order used here (contiguous, same as B packing): lane l (m = l&15, half = l>>4)
//   elems 0..15 -> K = kc + half*16 + e
// Fragment block = (m_tile * (K/32) + kc32); lane-major, 32 contiguous bytes
// per lane -> dst elem = tid*16. One thread per (m_tile, kc32, lane).
// ---------------------------------------------------------------------------
__global__ void __launch_bounds__(256) pack_a_kernel(
    const float* __restrict__ src, __bf16* __restrict__ dst, int K) {
  long tid = (long)blockIdx.x * blockDim.x + threadIdx.x;
  int  lane = (int)(tid & 31);
  long idx  = tid >> 5;
  int  kchunks = K >> 5;
  int  kc32 = (int)(idx % kchunks);
  long m_tl = idx / kchunks;
  long m    = m_tl * 16 + (lane & 15);
  int  half = lane >> 4;
  const float* s0 = src + m * (long)K + kc32 * 32 + half * 16;
  const float* s1 = s0 + 8;
  bf16x16 v;
#pragma unroll
  for (int e = 0; e < 8; ++e) v[e] = (__bf16)s0[e];
#pragma unroll
  for (int e = 0; e < 8; ++e) v[8 + e] = (__bf16)s1[e];
  *((bf16x16*)(dst + tid * 16)) = v;
}

// ---------------------------------------------------------------------------
// Pack f32 row-major [K,N] into bf16 WMMA B-fragments (32x16 tiles),
// split-half layout for conflict-minimal ds_load_b128 + straight async memcpy:
//   block (n_tile, kc): base = (n_tile*(K/32)+kc)*512 elems (1 KB)
//   half0 (e=0..7) at base + lane*8 ; half1 (e=8..15) at base + 256 + lane*8
// Lane l: n = n_tile*16 + (l&15), k(e) = kc*32 + (l>>4)*16 + e.
// One thread per (n_tile, kc, lane).
// ---------------------------------------------------------------------------
__global__ void __launch_bounds__(256) pack_b_kernel(
    const float* __restrict__ src, __bf16* __restrict__ dst, int K, int N) {
  long tid = (long)blockIdx.x * blockDim.x + threadIdx.x;
  int  lane = (int)(tid & 31);
  long idx  = tid >> 5;
  int  kchunks = K >> 5;
  int  kc32 = (int)(idx % kchunks);
  long n_tl = idx / kchunks;
  int  n    = (int)(n_tl * 16 + (lane & 15));
  int  k0   = kc32 * 32 + (lane >> 4) * 16;
  __bf16* base = dst + ((size_t)n_tl * kchunks + kc32) * 512;
  bf16x8 h0, h1;
#pragma unroll
  for (int e = 0; e < 8; ++e) h0[e] = (__bf16)src[(long)(k0 + e) * N + n];
#pragma unroll
  for (int e = 0; e < 8; ++e) h1[e] = (__bf16)src[(long)(k0 + 8 + e) * N + n];
  *(bf16x8*)(base + lane * 8)       = h0;
  *(bf16x8*)(base + 256 + lane * 8) = h1;
}

// ---------------------------------------------------------------------------
// Wneg[k, :] = -softmax(exp(lw[k, :])); one block (H threads) per row k.
// ---------------------------------------------------------------------------
__global__ void __launch_bounds__(512) softmax_negexp_kernel(
    const float* __restrict__ lw, float* __restrict__ wneg, int N) {
  __shared__ float red[512];
  int k = blockIdx.x, j = threadIdx.x;
  float v = expf(expf(lw[(long)k * N + j]));  // values bounded (~[1.8, 5])
  red[j] = v;
  __syncthreads();
  for (int s = 256; s > 0; s >>= 1) {
    if (j < s) red[j] += red[j + s];
    __syncthreads();
  }
  float inv = 1.0f / red[0];
  wneg[(long)k * N + j] = -v * inv;
}

// ---------------------------------------------------------------------------
// WMMA GEMM: C[M,N] = relu?( Apack @ Bpack + bias ), f32 out.
// Block = 256 threads = 8 waves; block tile 128(M) x 128(N).
// Waves: 4 along M x 2 along N; each wave owns 32x64 (2 A-frags x 4 B-frags
// -> 8 accumulators, 8 wmma per K-chunk).
// B panel staged in LDS (double-buffered 2 x 8 KB slices) via async
// global->LDS when available; A streamed with direct global_load_b128.
// ---------------------------------------------------------------------------
#if HAVE_ASYNC_LDS
#define ASYNC_CP(src_, dst_)                                                   \
  __builtin_amdgcn_global_load_async_to_lds_b128(                              \
      (__attribute__((address_space(1))) v4i_*)(src_),                         \
      (__attribute__((address_space(3))) v4i_*)(dst_), 0, 0)
#define ASYNC_WAIT(n_) __builtin_amdgcn_s_wait_asynccnt(n_)
#else
#define ASYNC_CP(src_, dst_) ((void)0)
#define ASYNC_WAIT(n_) ((void)0)
#endif

__global__ void __launch_bounds__(256) gemm_wmma_bf16(
    const __bf16* __restrict__ Ap, const __bf16* __restrict__ Bp,
    const float* __restrict__ bias, float* __restrict__ C,
    long M, int N, int K, int do_relu) {
  const int  kchunks = K >> 5;
  const int  lane = threadIdx.x & 31;
  const int  wave = threadIdx.x >> 5;
  const int  wm = wave & 3;                       // wave position along M
  const int  wn = wave >> 2;                      // wave position along N
  const long m_tile0 = (long)blockIdx.x * 8 + wm * 2;
  const int  n0_blk = blockIdx.y * 128;
  const int  ntl0 = wn * 4;                       // wave's local n-tile base
  const long ntbase = (long)(n0_blk >> 4);        // global n-tile base of block

  // 2 slices x (8 n-tiles x 512 bf16) = 16 KB
  __shared__ __align__(16) __bf16 ldsB[2 * 8 * 512];

  // Cooperative copy mapping: thread t -> (ntl = t>>5, lc = t&31), 32 B each.
  const int c_ntl = threadIdx.x >> 5;
  const int c_lc  = threadIdx.x & 31;

#define STAGE_B(kc_, buf_)                                                     \
  do {                                                                         \
    const __bf16* gb = Bp + ((size_t)(ntbase + c_ntl) * kchunks + (kc_)) * 512;\
    __bf16* lb = &ldsB[(buf_) * 4096 + c_ntl * 512];                           \
    if (HAVE_ASYNC_LDS) {                                                      \
      ASYNC_CP(gb + c_lc * 8, lb + c_lc * 8);                                  \
      ASYNC_CP(gb + 256 + c_lc * 8, lb + 256 + c_lc * 8);                      \
    } else {                                                                   \
      *(bf16x8*)(lb + c_lc * 8) = *(const bf16x8*)(gb + c_lc * 8);             \
      *(bf16x8*)(lb + 256 + c_lc * 8) =                                        \
          *(const bf16x8*)(gb + 256 + c_lc * 8);                               \
    }                                                                          \
  } while (0)

  const bf16x16* Af0 = (const bf16x16*)Ap + (m_tile0 + 0) * kchunks * 32 + lane;
  const bf16x16* Af1 = (const bf16x16*)Ap + (m_tile0 + 1) * kchunks * 32 + lane;

  f32x8 acc[2][4];
#pragma unroll
  for (int i = 0; i < 2; ++i)
#pragma unroll
    for (int j = 0; j < 4; ++j) acc[i][j] = (f32x8){};

  STAGE_B(0, 0);  // prologue: slice 0 -> buffer 0

  for (int kc = 0; kc < kchunks; ++kc) {
    const int cur = kc & 1;
    const bool more = (kc + 1) < kchunks;
    if (more) {
      STAGE_B(kc + 1, cur ^ 1);
      ASYNC_WAIT(2);  // this wave's `cur` slice copies done (2 newer in flight)
    } else {
      ASYNC_WAIT(0);  // drain
    }
    __syncthreads();  // everyone's `cur` slice visible

    bf16x16 a0 = Af0[(long)kc * 32];
    bf16x16 a1 = Af1[(long)kc * 32];
#pragma unroll
    for (int j = 0; j < 4; ++j) {
      const __bf16* lb = &ldsB[cur * 4096 + (ntl0 + j) * 512];
      Frag16 fb;
      fb.h[0] = *(const bf16x8*)(lb + lane * 8);
      fb.h[1] = *(const bf16x8*)(lb + 256 + lane * 8);
      acc[0][j] = __builtin_amdgcn_wmma_f32_16x16x32_bf16(
          false, a0, false, fb.v, (short)0, acc[0][j], false, false);
      acc[1][j] = __builtin_amdgcn_wmma_f32_16x16x32_bf16(
          false, a1, false, fb.v, (short)0, acc[1][j], false, false);
    }
    __syncthreads();  // done reading `cur` before it is restaged next iter
  }

  // C/D layout (ISA 7.12.2): lane l, VGPR r -> M = m0 + r + 8*(l>>4), N = (l&15)
  const int rb = (lane >> 4) * 8;
  const int nc = lane & 15;
#pragma unroll
  for (int i = 0; i < 2; ++i) {
    const long m0 = (m_tile0 + i) * 16;
#pragma unroll
    for (int j = 0; j < 4; ++j) {
      int   n  = n0_blk + (ntl0 + j) * 16 + nc;
      float bb = bias ? bias[n] : 0.0f;
#pragma unroll
      for (int r = 0; r < 8; ++r) {
        float val = acc[i][j][r] + bb;
        if (do_relu) val = fmaxf(val, 0.0f);
        C[(m0 + rb + r) * (long)N + n] = val;
      }
    }
  }
#undef STAGE_B
}

// ---------------------------------------------------------------------------
// FD layer: one lane per (b, h) channel, sequential scan over T with 6
// unrolled ODE substeps. 128-thread blocks so the 1024 wave32s spread over
// the whole chip (latency-chain bound). Loads coalesced over h per timestep.
// ---------------------------------------------------------------------------
__global__ void __launch_bounds__(128) fd_kernel(
    const float* __restrict__ in, const float* __restrict__ params,
    float* __restrict__ out_seq, float* __restrict__ out_last,
    int T, int H) {
  const int id = blockIdx.x * 128 + threadIdx.x;
  const int h = id % H;
  const int b = id / H;

  const float Ca_mu     = expf(params[0 * H + h]);
  const float inv_sig   = 1.0f / expf(params[1 * H + h]);
  const float inv_tauCa = 1.0f / expf(params[2 * H + h]);
  const float alpha     = expf(params[3 * H + h]);
  const float inv_tauE  = 1.0f / expf(params[4 * H + h]);
  const float beta      = expf(params[5 * H + h]);
  const float P_rel_max = 1.0f / (1.0f + expf(-params[6 * H + h]));
  const float k_min     = expf(params[7 * H + h]);
  const float k_del     = expf(params[8 * H + h]);
  const float dt = 1.0f / (float)ODE_STEPS;

  float EPSC = 0.0f, Ca = Ca_mu, R = 1.0f;
  const float* ip = in + (size_t)b * T * H + h;
  float cur = ip[0];

  for (int t = 0; t < T; ++t) {
    float nxt = (t + 1 < T) ? ip[(size_t)(t + 1) * H] : 0.0f;
#pragma unroll
    for (int s = 0; s < ODE_STEPS; ++s) {
      float Ca_diff = Ca - Ca_mu;
      float sg   = 1.0f / (1.0f + __expf(-Ca_diff * inv_sig));
      float temp = P_rel_max * sg * R * cur;
      EPSC = EPSC - dt * (EPSC * inv_tauE + beta * temp);
      float kr = k_min + k_del * sg;
      R  = R  + dt * (kr * (1.0f - R) - temp);
      Ca = Ca + dt * (alpha * cur - Ca_diff * inv_tauCa);
    }
    if (out_seq) out_seq[((size_t)b * T + t) * H + h] = EPSC;
    cur = nxt;
  }
  if (out_last) out_last[(size_t)b * H + h] = EPSC;
}

// ---------------------------------------------------------------------------
// Orchestration
// ---------------------------------------------------------------------------
extern "C" void kernel_launch(void* const* d_in, const int* in_sizes, int n_in,
                              void* d_out, int out_size, void* d_ws, size_t ws_size,
                              hipStream_t stream) {
  (void)in_sizes; (void)n_in; (void)out_size; (void)ws_size;

  const float* x      = (const float*)d_in[0];  // [64,1024,256]
  const float* lin_w  = (const float*)d_in[1];  // [256,512]
  const float* lin_b  = (const float*)d_in[2];  // [512]
  const float* fd1p   = (const float*)d_in[3];  // [9,512]
  const float* axon   = (const float*)d_in[4];  // [512,512]
  const float* fd2p   = (const float*)d_in[5];  // [9,512]
  float*       out    = (float*)d_out;          // [64,512]

  const int  Bn = 64, T = 1024, D = 256, H = 512;
  const long M  = (long)Bn * T;                 // 65536

  // Workspace layout (all segments 256B-aligned by construction)
  char* w = (char*)d_ws;
  __bf16* Apack1 = (__bf16*)w;  w += (size_t)M * D * 2;   // 32 MB
  __bf16* Bpack1 = (__bf16*)w;  w += (size_t)D * H * 2;   // 256 KB
  float*  Wneg   = (float*) w;  w += (size_t)H * H * 4;   // 1 MB
  __bf16* Bpack2 = (__bf16*)w;  w += (size_t)H * H * 2;   // 512 KB
  __bf16* Apack2 = (__bf16*)w;  w += (size_t)M * H * 2;   // 64 MB
  float*  hA     = (float*) w;  w += (size_t)M * H * 4;   // 128 MB (h1, then h3)
  float*  hB     = (float*) w;                            // 128 MB (h2)

  // 1) pack x -> A-fragments (bf16)
  {
    long threads = (M / 16) * (D / 32) * 32;  // 1,048,576
    pack_a_kernel<<<dim3((unsigned)(threads / 256)), dim3(256), 0, stream>>>(x, Apack1, D);
  }
  // 2) pack lin_w -> B-fragments
  {
    long threads = (H / 16) * (D / 32) * 32;  // 8192
    pack_b_kernel<<<dim3((unsigned)(threads / 256)), dim3(256), 0, stream>>>(lin_w, Bpack1, D, H);
  }
  // 3) Wneg = -softmax(exp(axon_log_w)) ; 4) pack -> B-fragments
  softmax_negexp_kernel<<<dim3(H), dim3(H), 0, stream>>>(axon, Wneg, H);
  {
    long threads = (H / 16) * (H / 32) * 32;  // 16384
    pack_b_kernel<<<dim3((unsigned)(threads / 256)), dim3(256), 0, stream>>>(Wneg, Bpack2, H, H);
  }
  // 5) h1 = relu(x @ lin_w + lin_b)
  gemm_wmma_bf16<<<dim3((unsigned)(M / 128), H / 128), dim3(256), 0, stream>>>(
      Apack1, Bpack1, lin_b, hA, M, H, D, 1);
  // 6) h2 = FD1(h1)
  fd_kernel<<<dim3((unsigned)(Bn * H / 128)), dim3(128), 0, stream>>>(
      hA, fd1p, hB, nullptr, T, H);
  // 7) pack h2 -> A-fragments
  {
    long threads = (M / 16) * (H / 32) * 32;  // 2,097,152
    pack_a_kernel<<<dim3((unsigned)(threads / 256)), dim3(256), 0, stream>>>(hB, Apack2, H);
  }
  // 8) h3 = h2 @ (-W)   (reuse hA buffer)
  gemm_wmma_bf16<<<dim3((unsigned)(M / 128), H / 128), dim3(256), 0, stream>>>(
      Apack2, Bpack2, nullptr, hA, M, H, H, 0);
  // 9) out = FD2(h3)[:, -1, :]
  fd_kernel<<<dim3((unsigned)(Bn * H / 128)), dim3(128), 0, stream>>>(
      hA, fd2p, nullptr, out, T, H);
}
